// RoPEMultiHeadAttention_14018773254764
// MI455X (gfx1250) — compile-verified, matrix operand order checked by reading, not run on hardware
//
#include <hip/hip_runtime.h>

// ---------------------------------------------------------------------------
// RoPE Multi-Head Attention for MI455X (gfx1250, wave32, WMMA bf16 + TDM)
//   x[2,2048,1024] f32, w_qkv[1024,3072] f32, w_out[1024,1024] f32 -> f32
// Pipeline: convert/transpose to bf16 | QKV GEMM (WMMA, double-buffered LDS,
// both tiles TDM-staged) | RoPE split | flash attention (WMMA) | out-proj.
// ---------------------------------------------------------------------------

#define D_MODEL 1024
#define NHEAD   16
#define HEAD_DIM 64
#define SEQ     2048
#define BATCH   2
#define ROWS    (BATCH * SEQ)   // 4096
#define QKV_N   (3 * D_MODEL)   // 3072

typedef unsigned int   u32;
typedef unsigned short u16;

typedef __attribute__((ext_vector_type(16))) __bf16 v16bf;
typedef __attribute__((ext_vector_type(8)))  float  v8f;
typedef __attribute__((ext_vector_type(4)))  u32    u32x4;
typedef __attribute__((ext_vector_type(2)))  u32    u32x2;
typedef __attribute__((ext_vector_type(4)))  float  f32x4;
typedef __attribute__((ext_vector_type(8)))  int    i32x8;
typedef __attribute__((ext_vector_type(4)))  int    i32x4;

// Tensor Data Mover availability (device pass only; host pass takes fallback)
#if defined(__has_builtin)
#if __has_builtin(__builtin_amdgcn_tensor_load_to_lds) && \
    __has_builtin(__builtin_amdgcn_s_wait_tensorcnt)
#define USE_TDM 1
#endif
#endif

// WMMA 16x16x32 bf16 fragment: 8 VGPRs = two 16-byte runs per lane.
union FragU {
  struct { u32x4 lo, hi; } q;
  v16bf v;
};

__device__ __forceinline__ u16 f2bf(float f) {           // RNE f32 -> bf16
  u32 u = __float_as_uint(f);
  u32 r = 0x7FFFu + ((u >> 16) & 1u);
  return (u16)((u + r) >> 16);
}
__device__ __forceinline__ u32 pack2(float a, float b) {
  return (u32)f2bf(a) | ((u32)f2bf(b) << 16);
}
__device__ __forceinline__ v8f v8f_zero() {
  v8f z = {0.f, 0.f, 0.f, 0.f, 0.f, 0.f, 0.f, 0.f};
  return z;
}

// ---------------------------------------------------------------------------
// Contiguous f32 -> bf16 convert (vector of 4 per thread)
// ---------------------------------------------------------------------------
__global__ void k_convert_bf16(const float* __restrict__ src,
                               u16* __restrict__ dst, long total) {
  long idx = ((long)blockIdx.x * blockDim.x + threadIdx.x) * 4;
  if (idx >= total) return;
  f32x4 f = *(const f32x4*)(src + idx);
  u32x2 p;
  p.x = pack2(f.x, f.y);
  p.y = pack2(f.z, f.w);
  *(u32x2*)(dst + idx) = p;
}

// ---------------------------------------------------------------------------
// W[k][n] f32 (row-major) -> Wt[n][k] bf16 (K-major, i.e. "transposed B")
// ---------------------------------------------------------------------------
__global__ void k_transpose_bf16(const float* __restrict__ w,
                                 u16* __restrict__ wt, int K, int N) {
  long idx = (long)blockIdx.x * blockDim.x + threadIdx.x;
  long total = (long)K * N;
  if (idx >= total) return;
  int k = (int)(idx / N);
  int n = (int)(idx % N);
  wt[(long)n * K + k] = f2bf(w[idx]);
}

// ---------------------------------------------------------------------------
// GEMM: C[M,N] f32 = A[M,K] bf16  x  Bt[N,K] bf16
// Block 256 thr = 8 waves; tile 128x128x32; wave computes 32x64 (8 WMMA/step).
// Double-buffered LDS, one barrier per k-step. Both tiles DMA'd by the
// Tensor Data Mover: each wave issues two TENSOR_LOAD_TO_LDS (its 16-row
// slice of A and of B). D#.pad (16 DW interval, +12 DW) gives the padded
// 56-element LDS row stride the b128 fragment loads expect (16B-aligned,
// conflict-free bank pattern).
// ---------------------------------------------------------------------------
#define BK    32
#define LDST  56

#ifdef USE_TDM
// Issue TENSOR_LOAD_TO_LDS for a [16 rows x 32 cols] bf16 tile of
// S[nrows][ld] starting at (row0, k0) into LDS at lds_dst.
__device__ __forceinline__ void tdm_stage_tile(const u16* __restrict__ S,
                                               int nrows, int ld,
                                               int row0, int k0,
                                               const u16* lds_dst) {
  unsigned long long ga =
      (unsigned long long)(size_t)(S + (long)row0 * ld + k0);
  // --- D# group 0: count=1 | lds_addr | global_addr | type=2 ---
  u32x4 g0;
  g0.x = 0x1u;                              // count=1, user mode
  g0.y = (u32)(size_t)lds_dst;              // flat[31:0] == LDS byte offset
  g0.z = (u32)ga;                           // global_addr[31:0]
  g0.w = (u32)(ga >> 32) | (2u << 30);      // global_addr[56:32] | type=2
  // --- D# group 1 ---
  // w0: data_size=2B (1<<16) | pad_enable (1<<20) | pad_interval=16DW (3<<22)
  //     | pad_amount=12DW (11<<25)  -> LDS row stride 16+12 = 28 DW = 56 bf16
  i32x8 g1;
  g1[0] = (int)((1u << 16) | (1u << 20) | (3u << 22) | (11u << 25));
  g1[1] = (int)(((u32)ld & 0xFFFFu) << 16);               // tensor_dim0[15:0]
  g1[2] = (int)((((u32)ld >> 16) & 0xFFFFu) |             // tensor_dim0[31:16]
                (((u32)nrows & 0xFFFFu) << 16));          // tensor_dim1[15:0]
  g1[3] = (int)((((u32)nrows >> 16) & 0xFFFFu) |          // tensor_dim1[31:16]
                (32u << 16));                             // tile_dim0 = 32
  g1[4] = 16;                                 // tile_dim1 = 16, tile_dim2 = 0
  g1[5] = (int)(u32)ld;                       // tensor_dim0_stride[31:0]
  g1[6] = 0;                                  // stride hi, dim1_stride
  g1[7] = 0;
  i32x4 z4 = {0, 0, 0, 0};
#if __clang_major__ >= 23
  i32x8 z8 = {0, 0, 0, 0, 0, 0, 0, 0};
  __builtin_amdgcn_tensor_load_to_lds(g0, g1, z4, z4, z8, 0);
#else
  __builtin_amdgcn_tensor_load_to_lds(g0, g1, z4, z4, 0);
#endif
}
#else
// Fallback: cooperative 128x32 bf16 tile staging via b128 load + ds_store.
__device__ __forceinline__ void stage_tile_128x32(const u16* __restrict__ S,
                                                  int ld, int row0, int k0,
                                                  u16* buf, int tid) {
#pragma unroll
  for (int i = 0; i < 2; ++i) {
    int idx = tid + i * 256;                 // 0..511 b128 chunks
    int n = idx >> 2;
    int c = idx & 3;
    u32x4 u = *(const u32x4*)(S + (long)(row0 + n) * ld + k0 + c * 8);
    *(u32x4*)&buf[n * LDST + c * 8] = u;
  }
}
#endif

__global__ __launch_bounds__(256)
void k_gemm_bf16(const u16* __restrict__ A, const u16* __restrict__ Bt,
                 float* __restrict__ C, int M, int N, int K) {
  __shared__ __attribute__((aligned(16))) u16 ldsA[2][128 * LDST];
  __shared__ __attribute__((aligned(16))) u16 ldsB[2][128 * LDST];

  const int tid  = threadIdx.x;
  const int lane = tid & 31;
  const int wave = tid >> 5;
  const int wm   = wave & 3;    // 32-row band
  const int wn   = wave >> 2;   // 64-col band
  const int l15  = lane & 15;   // A: row m / B: col n
  const int kg   = lane >> 4;   // K half selector
  const int r0   = blockIdx.y * 128;
  const int c0   = blockIdx.x * 128;
  const int nsteps = K / BK;

  v8f acc[8];
#pragma unroll
  for (int i = 0; i < 8; ++i) acc[i] = v8f_zero();

  // ---- prologue: stage tile 0 into buffer 0 ----
#ifdef USE_TDM
  const int wv = __builtin_amdgcn_readfirstlane((int)(threadIdx.x >> 5));
  tdm_stage_tile(A,  M, K, r0 + wv * 16, 0, &ldsA[0][wv * 16 * LDST]);
  tdm_stage_tile(Bt, N, K, c0 + wv * 16, 0, &ldsB[0][wv * 16 * LDST]);
#else
  stage_tile_128x32(A,  K, r0, 0, ldsA[0], tid);
  stage_tile_128x32(Bt, K, c0, 0, ldsB[0], tid);
#endif

  int cur = 0;
  for (int kt = 0; kt < nsteps; ++kt) {
#ifdef USE_TDM
    __builtin_amdgcn_s_wait_tensorcnt(0);    // this wave's TDMs landed in LDS
#endif
    __syncthreads();                         // buf[cur] visible to all waves

    // ---- fragment loads from buf[cur] ----
    FragU a[2], b[4];
#pragma unroll
    for (int ms = 0; ms < 2; ++ms) {
      const u16* p = &ldsA[cur][(wm * 32 + ms * 16 + l15) * LDST + kg * 8];
      a[ms].q.lo = *(const u32x4*)p;
      a[ms].q.hi = *(const u32x4*)(p + 16);
    }
#pragma unroll
    for (int ns = 0; ns < 4; ++ns) {
      const u16* p = &ldsB[cur][(wn * 64 + ns * 16 + l15) * LDST + kg * 8];
      b[ns].q.lo = *(const u32x4*)p;
      b[ns].q.hi = *(const u32x4*)(p + 16);
    }

    // ---- stage next tile into buf[cur^1] (overlaps with WMMA below) ----
    if (kt + 1 < nsteps) {
      const int k1 = (kt + 1) * BK;
#ifdef USE_TDM
      tdm_stage_tile(A,  M, K, r0 + wv * 16, k1,
                     &ldsA[cur ^ 1][wv * 16 * LDST]);
      tdm_stage_tile(Bt, N, K, c0 + wv * 16, k1,
                     &ldsB[cur ^ 1][wv * 16 * LDST]);
#else
      stage_tile_128x32(A,  K, r0, k1, ldsA[cur ^ 1], tid);
      stage_tile_128x32(Bt, K, c0, k1, ldsB[cur ^ 1], tid);
#endif
    }

    // ---- 8 WMMAs ----
#pragma unroll
    for (int ms = 0; ms < 2; ++ms)
#pragma unroll
      for (int ns = 0; ns < 4; ++ns)
        acc[ms * 4 + ns] = __builtin_amdgcn_wmma_f32_16x16x32_bf16(
            false, a[ms].v, false, b[ns].v, (short)0, acc[ms * 4 + ns],
            false, false);
    cur ^= 1;
  }

  // epilogue: C layout = row (lane>>4)*8+r, col lane&15
  const int mg = lane >> 4;
#pragma unroll
  for (int ms = 0; ms < 2; ++ms)
#pragma unroll
    for (int ns = 0; ns < 4; ++ns)
#pragma unroll
      for (int r = 0; r < 8; ++r) {
        int row = r0 + wm * 32 + ms * 16 + mg * 8 + r;
        int col = c0 + wn * 64 + ns * 16 + l15;
        C[(long)row * N + col] = acc[ms * 4 + ns][r];
      }
  (void)M;
}

// ---------------------------------------------------------------------------
// RoPE + head split: qkv[4096][3072] f32 ->
//   Q,K bf16 [b,h,t,64]   (rows contiguous: direct A/B fragment loads)
//   Vt  bf16 [b,h,64,t]   (transposed: PV B-fragments contiguous)
// ---------------------------------------------------------------------------
__global__ void k_rope_split(const float* __restrict__ qkv,
                             u16* __restrict__ Q, u16* __restrict__ Kc,
                             u16* __restrict__ Vt) {
  int idx = blockIdx.x * blockDim.x + threadIdx.x;  // 2^21 threads
  int i = idx & 31;
  int h = (idx >> 5) & (NHEAD - 1);
  int t = (idx >> 9) & (SEQ - 1);
  int b = idx >> 20;
  if (b >= BATCH) return;

  const float* base = qkv + ((long)b * SEQ + t) * QKV_N + h * HEAD_DIM;
  float q0 = base[2 * i],               q1 = base[2 * i + 1];
  float k0 = base[D_MODEL + 2 * i],     k1 = base[D_MODEL + 2 * i + 1];
  float v0 = base[2 * D_MODEL + 2 * i], v1 = base[2 * D_MODEL + 2 * i + 1];

  // emb = concat(freqs, freqs): index j uses inv_freq[j & 31]
  float inv0 = __powf(10000.0f, -(float)((2 * i) & 31) / 32.0f);
  float inv1 = __powf(10000.0f, -(float)((2 * i + 1) & 31) / 32.0f);
  float s0, c0, s1, c1;
  __sincosf((float)t * inv0, &s0, &c0);
  __sincosf((float)t * inv1, &s1, &c1);

  float qo0 = q0 * c0 - q1 * s0;
  float qo1 = q1 * c1 + q0 * s1;
  float ko0 = k0 * c0 - k1 * s0;
  float ko1 = k1 * c1 + k0 * s1;

  long hb = (long)b * NHEAD + h;
  long qko = (hb * SEQ + t) * HEAD_DIM + 2 * i;
  Q[qko]      = f2bf(qo0);  Q[qko + 1]  = f2bf(qo1);
  Kc[qko]     = f2bf(ko0);  Kc[qko + 1] = f2bf(ko1);
  long vto = (hb * HEAD_DIM + 2 * i) * SEQ + t;
  Vt[vto]       = f2bf(v0);
  Vt[vto + SEQ] = f2bf(v1);
}

// ---------------------------------------------------------------------------
// Flash attention, 1 wave per 16-row Q tile (4 waves/block).
// Per 32-key chunk: 4 score WMMA + online softmax (shfl_xor row reductions
// within the 16-lane C-matrix halves) + P relayout via LDS + 4 PV WMMA.
// Output y written directly as bf16 (A-operand of the out-projection GEMM).
// ---------------------------------------------------------------------------
#define PST 56   // P-tile LDS stride (bf16): 112 B, 16B-aligned frag loads

__global__ __launch_bounds__(128)
void k_flash_attn(const u16* __restrict__ Q, const u16* __restrict__ Kc,
                  const u16* __restrict__ Vt, u16* __restrict__ Yb) {
  __shared__ __attribute__((aligned(16))) u16 ldsP[4][16 * PST];

  const int lane = threadIdx.x & 31;
  const int wave = threadIdx.x >> 5;
  const int h  = blockIdx.y;
  const int b  = blockIdx.z;
  const int qt = blockIdx.x * 4 + wave;
  const int t0 = qt * 16;
  const int l15 = lane & 15;
  const int kg  = lane >> 4;

  const long hb = (long)b * NHEAD + h;
  const u16* Qb = Q  + hb * SEQ * HEAD_DIM;
  const u16* Kb = Kc + hb * SEQ * HEAD_DIM;
  const u16* Vb = Vt + hb * HEAD_DIM * SEQ;
  u16* myP = ldsP[wave];

  // Q fragments (16x64 -> 2 k-steps), loaded once from global
  FragU aQ[2];
#pragma unroll
  for (int f = 0; f < 2; ++f) {
    const u16* p = Qb + (long)(t0 + l15) * HEAD_DIM + f * 32 + kg * 8;
    aQ[f].q.lo = *(const u32x4*)p;
    aQ[f].q.hi = *(const u32x4*)(p + 16);
  }

  v8f acc[4];
#pragma unroll
  for (int i = 0; i < 4; ++i) acc[i] = v8f_zero();
  float mrun[8], lrun[8];
#pragma unroll
  for (int r = 0; r < 8; ++r) { mrun[r] = -1e30f; lrun[r] = 0.f; }

  const float sscale = 0.125f;  // 1/sqrt(64)

  for (int kt = 0; kt < SEQ / 32; ++kt) {
    // prefetch next chunk's K/V lines (global_prefetch_b8)
    if (kt + 1 < SEQ / 32) {
      __builtin_prefetch(Kb + (long)((kt + 1) * 32 + l15) * HEAD_DIM + kg * 32,
                         0, 3);
      __builtin_prefetch(Vb + (long)(l15) * SEQ + (kt + 1) * 32 + kg * 8,
                         0, 3);
    }

    // ---- scores: two 16x16 tiles covering 32 keys ----
    v8f s[2];
#pragma unroll
    for (int sub = 0; sub < 2; ++sub) {
      const int kb = kt * 32 + sub * 16;
      FragU bK[2];
#pragma unroll
      for (int f = 0; f < 2; ++f) {
        const u16* p = Kb + (long)(kb + l15) * HEAD_DIM + f * 32 + kg * 8;
        bK[f].q.lo = *(const u32x4*)p;
        bK[f].q.hi = *(const u32x4*)(p + 16);
      }
      v8f sv = v8f_zero();
      sv = __builtin_amdgcn_wmma_f32_16x16x32_bf16(false, aQ[0].v, false,
                                                   bK[0].v, (short)0, sv,
                                                   false, false);
      sv = __builtin_amdgcn_wmma_f32_16x16x32_bf16(false, aQ[1].v, false,
                                                   bK[1].v, (short)0, sv,
                                                   false, false);
      s[sub] = sv * sscale;
    }

    // ---- online softmax (row r lives at vgpr r across 16-lane half) ----
    float pr0[8], pr1[8];
#pragma unroll
    for (int r = 0; r < 8; ++r) {
      float rm = fmaxf(s[0][r], s[1][r]);
#pragma unroll
      for (int off = 8; off; off >>= 1) rm = fmaxf(rm, __shfl_xor(rm, off));
      float mn   = fmaxf(mrun[r], rm);
      float corr = __expf(mrun[r] - mn);
      float p0 = __expf(s[0][r] - mn);
      float p1 = __expf(s[1][r] - mn);
      float sum = p0 + p1;
#pragma unroll
      for (int off = 8; off; off >>= 1) sum += __shfl_xor(sum, off);
      lrun[r] = lrun[r] * corr + sum;
      mrun[r] = mn;
#pragma unroll
      for (int nb = 0; nb < 4; ++nb) acc[nb][r] *= corr;
      pr0[r] = p0; pr1[r] = p1;
    }

    // ---- relayout P (C-format -> A-fragment format) via private LDS ----
#pragma unroll
    for (int r = 0; r < 8; ++r) {
      myP[(kg * 8 + r) * PST + l15]      = f2bf(pr0[r]);
      myP[(kg * 8 + r) * PST + 16 + l15] = f2bf(pr1[r]);
    }
    FragU aP;
    {
      const u16* p = &myP[l15 * PST + kg * 8];
      aP.q.lo = *(const u32x4*)p;
      aP.q.hi = *(const u32x4*)(p + 16);
    }

    // ---- PV: O(16x64) += P(16x32) x V(32x64) ----
#pragma unroll
    for (int nb = 0; nb < 4; ++nb) {
      FragU bV;
      const u16* p = Vb + (long)(nb * 16 + l15) * SEQ + kt * 32 + kg * 8;
      bV.q.lo = *(const u32x4*)p;
      bV.q.hi = *(const u32x4*)(p + 16);
      acc[nb] = __builtin_amdgcn_wmma_f32_16x16x32_bf16(
          false, aP.v, false, bV.v, (short)0, acc[nb], false, false);
    }
  }

  // ---- epilogue: y[b][t][h*64 + d] as bf16 (feeds out-proj GEMM) ----
#pragma unroll
  for (int r = 0; r < 8; ++r) {
    float inv = 1.0f / lrun[r];
    int t = t0 + kg * 8 + r;
    long rowoff = ((long)b * SEQ + t) * D_MODEL + h * HEAD_DIM;
#pragma unroll
    for (int nb = 0; nb < 4; ++nb)
      Yb[rowoff + nb * 16 + l15] = f2bf(acc[nb][r] * inv);
  }
}

// ---------------------------------------------------------------------------
// Launcher
// ---------------------------------------------------------------------------
extern "C" void kernel_launch(void* const* d_in, const int* in_sizes, int n_in,
                              void* d_out, int out_size, void* d_ws,
                              size_t ws_size, hipStream_t stream) {
  (void)in_sizes; (void)n_in; (void)out_size; (void)ws_size;
  const float* x     = (const float*)d_in[0];
  const float* w_qkv = (const float*)d_in[1];
  const float* w_out = (const float*)d_in[2];
  float* out = (float*)d_out;

  char* ws = (char*)d_ws;
  size_t off = 0;
  auto alloc = [&](size_t bytes) -> void* {
    void* p = ws + off;
    off = (off + bytes + 255) & ~(size_t)255;
    return p;
  };

  u16*   xb    = (u16*)  alloc((size_t)ROWS * D_MODEL * sizeof(u16));    //  8 MB
  u16*   WqkvT = (u16*)  alloc((size_t)QKV_N * D_MODEL * sizeof(u16));   //  6 MB
  u16*   WoutT = (u16*)  alloc((size_t)D_MODEL * D_MODEL * sizeof(u16)); //  2 MB
  float* qkv   = (float*)alloc((size_t)ROWS * QKV_N * sizeof(float));    // 48 MB
  u16*   Qb    = (u16*)  alloc((size_t)ROWS * D_MODEL * sizeof(u16));    //  8 MB
  u16*   Kb    = (u16*)  alloc((size_t)ROWS * D_MODEL * sizeof(u16));    //  8 MB
  u16*   Vt    = (u16*)  alloc((size_t)ROWS * D_MODEL * sizeof(u16));    //  8 MB
  u16*   yb    = (u16*)  alloc((size_t)ROWS * D_MODEL * sizeof(u16));    //  8 MB

  {
    long total = (long)ROWS * D_MODEL;
    k_convert_bf16<<<(int)(total / 4 / 256), 256, 0, stream>>>(x, xb, total);
  }
  {
    long total = (long)D_MODEL * QKV_N;
    k_transpose_bf16<<<(int)((total + 255) / 256), 256, 0, stream>>>(
        w_qkv, WqkvT, D_MODEL, QKV_N);
  }
  {
    long total = (long)D_MODEL * D_MODEL;
    k_transpose_bf16<<<(int)((total + 255) / 256), 256, 0, stream>>>(
        w_out, WoutT, D_MODEL, D_MODEL);
  }
  k_gemm_bf16<<<dim3(QKV_N / 128, ROWS / 128), 256, 0, stream>>>(
      xb, WqkvT, qkv, ROWS, QKV_N, D_MODEL);
  k_rope_split<<<(ROWS * NHEAD * 32) / 256, 256, 0, stream>>>(qkv, Qb, Kb, Vt);
  k_flash_attn<<<dim3(SEQ / 64, NHEAD, BATCH), 128, 0, stream>>>(
      Qb, Kb, Vt, yb);
  k_gemm_bf16<<<dim3(D_MODEL / 128, ROWS / 128), 256, 0, stream>>>(
      yb, WoutT, out, ROWS, D_MODEL, D_MODEL);
}
